// SectorAttentionV2_2628519985692
// MI455X (gfx1250) — matrix operand order repaired and checked by reading, not compiled, not run on hardware
//
#include <hip/hip_runtime.h>
#include <hip/hip_bf16.h>

typedef __attribute__((ext_vector_type(16))) __bf16 v16bf;
typedef __attribute__((ext_vector_type(8)))  float  v8f;

// ---- problem dims (fixed by reference) ----
constexpr int Bb   = 2;
constexpr int HRES = 4;
constexpr int WRES = 8192;
constexpr int Cc   = 512;
constexpr int NHh  = 8;
constexpr int HDd  = 64;               // Cc / NHh
constexpr int Ll   = HRES * WRES;      // 32768
constexpr int NSs  = 4 * WRES;         // 32768
constexpr int NWw  = Bb * WRES;        // 16384
constexpr float SCALE = 0.125f;        // 64^-0.5
constexpr float EPSb  = 1e-5f;

// ---- gfx1250 async-copy helpers (ASYNCcnt-tracked global->LDS DMA) ----
__device__ __forceinline__ void async_copy_b128(uint32_t lds_byte_off, const void* gaddr) {
    asm volatile("global_load_async_to_lds_b128 %0, %1, off"
                 :: "v"(lds_byte_off), "v"(gaddr) : "memory");
}
__device__ __forceinline__ void wait_asynccnt0() {
    asm volatile("s_wait_asynccnt 0x0" ::: "memory");
}
__device__ __forceinline__ uint32_t lds_offset_of(const void* p) {
    // flat LDS aperture keeps the byte offset in addr[31:0]
    return (uint32_t)(uintptr_t)p;
}

// ============================================================
// 0) Weight prep: WT[n][k] = bf16(W[k][n])   (512x512, once per weight)
// ============================================================
__global__ __launch_bounds__(256)
void transpose_w_kernel(const float* __restrict__ W, __bf16* __restrict__ WT)
{
    int gid = blockIdx.x * blockDim.x + threadIdx.x;   // over 512*512
    int n = gid >> 9, k = gid & 511;
    WT[(size_t)n * Cc + k] = (__bf16)W[(size_t)k * Cc + n];
}

// ============================================================
// 1) Relative position bias: [NW, NH, 4, 4]  (one thread/window)
// ============================================================
__global__ __launch_bounds__(128)
void bias_kernel(const float* __restrict__ s_pos, const float* __restrict__ x_pos,
                 const float* __restrict__ pw1,  const float* __restrict__ pb1,
                 const float* __restrict__ bn_g, const float* __restrict__ bn_b,
                 const float* __restrict__ bn_m, const float* __restrict__ bn_v,
                 const float* __restrict__ pw2,  const float* __restrict__ pb2,
                 float* __restrict__ bias)
{
    int n = blockIdx.x * blockDim.x + threadIdx.x;
    if (n >= NWw) return;
    int b = n / WRES, w = n % WRES;

    float xp0[4], xp1[4], sp0[4], sp1[4];
    #pragma unroll
    for (int i = 0; i < 4; ++i) {
        size_t l = (size_t)i * WRES + w;
        xp0[i] = x_pos[((size_t)b * Ll + l) * 2 + 0];
        xp1[i] = x_pos[((size_t)b * Ll + l) * 2 + 1];
    }
    #pragma unroll
    for (int j = 0; j < 4; ++j) {
        sp0[j] = s_pos[(((size_t)b * 4 + j) * WRES + w) * 2 + 0];
        sp1[j] = s_pos[(((size_t)b * 4 + j) * WRES + w) * 2 + 1];
    }

    float w0[16], w1[16], c1[16], sc[16], sh[16];
    #pragma unroll
    for (int o = 0; o < 16; ++o) {
        w0[o] = pw1[2 * o + 0];
        w1[o] = pw1[2 * o + 1];
        c1[o] = pb1[o];
        float s_ = bn_g[o] * rsqrtf(bn_v[o] + EPSb);
        sc[o] = s_;
        sh[o] = bn_b[o] - bn_m[o] * s_;
    }

    #pragma unroll
    for (int i = 0; i < 4; ++i) {
        #pragma unroll
        for (int j = 0; j < 4; ++j) {
            float r0 = xp0[i] - sp0[j];
            float r1 = xp1[i] - sp1[j];
            float hv[16];
            #pragma unroll
            for (int o = 0; o < 16; ++o) {
                float h = fmaf(w0[o], r0, fmaf(w1[o], r1, c1[o]));
                h = fmaf(h, sc[o], sh[o]);
                hv[o] = fmaxf(h, 0.0f);
            }
            #pragma unroll
            for (int hh = 0; hh < NHh; ++hh) {
                float acc = pb2[hh];
                #pragma unroll
                for (int o = 0; o < 16; ++o)
                    acc = fmaf(pw2[hh * 16 + o], hv[o], acc);
                bias[(((size_t)n * NHh + hh) * 16) + i * 4 + j] = acc;
            }
        }
    }
}

// ============================================================
// 2) Projection GEMM (single or fused-dual output):
//    O(bf16)[M,512] = In(f32)[M,512] @ W(f32->bf16)[512,512] + b
//    Block tile 128x64, 8 waves, wave = 2x2 of 16x16x32 bf16 WMMA.
//    A: f32 global -> cvt_pk_bf16 -> padded LDS (double-buffered).
//    B: pre-transposed bf16 WT -> global_load_async_to_lds_b128.
// ============================================================
constexpr int BM = 128, BN = 64, BK = 32;
constexpr int LDA = 40;   // padded bf16 row stride (80 B, 16B-aligned, conflict-free)
constexpr int LDB = 40;
constexpr int NT  = Cc / BK;   // 16 K-steps

template<bool DUAL>
__global__ __launch_bounds__(256)
void gemm_proj_kernel(const float* __restrict__ A,
                      const __bf16* __restrict__ WT1, const float* __restrict__ b1,
                      __bf16* __restrict__ O1,
                      const __bf16* __restrict__ WT2, const float* __restrict__ b2,
                      __bf16* __restrict__ O2)
{
    __shared__ __bf16 sA [2][BM * LDA];   // 2 x 10 KB
    __shared__ __bf16 sB1[2][BN * LDB];   // 2 x 5 KB
    __shared__ __bf16 sB2[2][BN * LDB];   // 2 x 5 KB (only touched when DUAL)

    const int tid  = threadIdx.x;
    const int lane = tid & 31;
    const int wave = tid >> 5;
    const int wm   = wave & 3;   // M sub-tile: wm*32
    const int wn   = wave >> 2;  // N sub-tile: wn*32

    const int nBlocksN   = Cc / BN;   // 8
    const size_t rowBase = (size_t)(blockIdx.x / nBlocksN) * BM;
    const int    colBase = (blockIdx.x % nBlocksN) * BN;

    // staging roles
    const int arow = tid >> 1;             // A: 2 threads per row, 64 B each
    const int acol = (tid & 1) * 16;       //    16 f32 per thread
    const int bcol = tid >> 2;             // B: 4 chunks (16 B) per 64-B col-row
    const int bchk = (tid & 3) * 8;        //    chunk offset in bf16 elems

    auto issueB = [&](int k0, int nb) {
        async_copy_b128(lds_offset_of(&sB1[nb][bcol * LDB + bchk]),
                        &WT1[(size_t)(colBase + bcol) * Cc + k0 + bchk]);
        if constexpr (DUAL)
            async_copy_b128(lds_offset_of(&sB2[nb][bcol * LDB + bchk]),
                            &WT2[(size_t)(colBase + bcol) * Cc + k0 + bchk]);
    };

    float2 ra[8];
    auto loadA = [&](int k0) {
        #pragma unroll
        for (int it = 0; it < 8; ++it)
            ra[it] = *reinterpret_cast<const float2*>(
                &A[(rowBase + arow) * Cc + k0 + acol + it * 2]);
    };
    auto storeA = [&](int nb) {
        #pragma unroll
        for (int it = 0; it < 8; ++it) {
            sA[nb][arow * LDA + acol + it * 2 + 0] = (__bf16)ra[it].x;
            sA[nb][arow * LDA + acol + it * 2 + 1] = (__bf16)ra[it].y;
        }
    };

    v8f acc1[2][2] = {};
    v8f acc2[2][2] = {};

    // prologue: stage tile 0 into buffer 0
    issueB(0, 0);
    loadA(0);
    storeA(0);
    wait_asynccnt0();
    __syncthreads();

    const int khalf = lane >> 4;
    const int rlo   = lane & 15;

    for (int kt = 0; kt < NT; ++kt) {
        const int cur = kt & 1;
        if (kt + 1 < NT) {
            issueB((kt + 1) * BK, cur ^ 1);
            loadA((kt + 1) * BK);
        }
        if (kt + 2 < NT)   // warm L2 for the tile after next (global_prefetch_b8)
            __builtin_prefetch(&A[(rowBase + arow) * Cc + (kt + 2) * BK + acol], 0, 1);

        // fragments: all contiguous 16-B runs in padded LDS -> ds_load_b128
        v16bf afrag[2], bfrag1[2], bfrag2[2];
        #pragma unroll
        for (int sm = 0; sm < 2; ++sm) {
            int base = (wm * 32 + sm * 16 + rlo) * LDA;
            #pragma unroll
            for (int t = 0; t < 8; ++t) {
                int kk = 16 * khalf + 2 * t;
                afrag[sm][2 * t + 0] = sA[cur][base + kk + 0];
                afrag[sm][2 * t + 1] = sA[cur][base + kk + 1];
            }
        }
        #pragma unroll
        for (int sn = 0; sn < 2; ++sn) {
            int base = (wn * 32 + sn * 16 + rlo) * LDB + khalf * 16;
            #pragma unroll
            for (int e = 0; e < 16; ++e) {
                bfrag1[sn][e] = sB1[cur][base + e];
                if constexpr (DUAL) bfrag2[sn][e] = sB2[cur][base + e];
            }
        }

        #pragma unroll
        for (int sm = 0; sm < 2; ++sm)
            #pragma unroll
            for (int sn = 0; sn < 2; ++sn) {
                acc1[sm][sn] = __builtin_amdgcn_wmma_f32_16x16x32_bf16(
                    false, afrag[sm], false, bfrag1[sn],
                    (short)0, acc1[sm][sn], false, false);
                if constexpr (DUAL)
                    acc2[sm][sn] = __builtin_amdgcn_wmma_f32_16x16x32_bf16(
                        false, afrag[sm], false, bfrag2[sn],
                        (short)0, acc2[sm][sn], false, false);
            }

        if (kt + 1 < NT) {
            storeA(cur ^ 1);
            wait_asynccnt0();
        }
        __syncthreads();
    }

    // epilogue: bias add, convert, store (C layout: N=lane&15, M=8*(lane>>4)+r)
    const int nlo = lane & 15;
    const int mhi = (lane >> 4) * 8;
    #pragma unroll
    for (int sm = 0; sm < 2; ++sm) {
        #pragma unroll
        for (int sn = 0; sn < 2; ++sn) {
            int col = colBase + wn * 32 + sn * 16 + nlo;
            float bv1 = b1[col];
            float bv2 = DUAL ? b2[col] : 0.0f;
            #pragma unroll
            for (int r = 0; r < 8; ++r) {
                size_t row = rowBase + wm * 32 + sm * 16 + mhi + r;
                O1[row * Cc + col] = (__bf16)(acc1[sm][sn][r] + bv1);
                if constexpr (DUAL)
                    O2[row * Cc + col] = (__bf16)(acc2[sm][sn][r] + bv2);
            }
        }
    }
}

// ============================================================
// 3) Windowed sector attention (raw-view k/v gather), one thread per (window, head)
// ============================================================
__global__ __launch_bounds__(256)
void attn_kernel(const __bf16* __restrict__ Q, const __bf16* __restrict__ K,
                 const __bf16* __restrict__ V, const float* __restrict__ bias,
                 float* __restrict__ out)
{
    int gid = blockIdx.x * blockDim.x + threadIdx.x;   // NW*NH threads
    if (gid >= NWw * NHh) return;
    int hh = gid / NWw;             // head-major: lanes span consecutive windows
    int n  = gid % NWw;
    int b  = n / WRES, w = n % WRES;
    int w9 = w >> 9, cK = w & 511;  // raw-view decomposition

    float attn[4][4];
    #pragma unroll
    for (int i = 0; i < 4; ++i)
        #pragma unroll
        for (int j = 0; j < 4; ++j)
            attn[i][j] = bias[(((size_t)n * NHh + hh) * 16) + i * 4 + j];

    const size_t qbase = (size_t)b * Ll * Cc + hh * HDd;
    const size_t kbase = (size_t)b * NSs * Cc + cK;

    for (int d = 0; d < HDd; ++d) {
        float qv[4], kv[4];
        #pragma unroll
        for (int i = 0; i < 4; ++i)
            qv[i] = SCALE * (float)Q[qbase + (size_t)(i * WRES + w) * Cc + d];
        int nsb = (hh * HDd + d) * 64 + w9;   // ns = (hh*64+d)*64 + hs*16 + (w>>9)
        #pragma unroll
        for (int j = 0; j < 4; ++j)
            kv[j] = (float)K[kbase + (size_t)(nsb + j * 16) * Cc];
        #pragma unroll
        for (int i = 0; i < 4; ++i)
            #pragma unroll
            for (int j = 0; j < 4; ++j)
                attn[i][j] = fmaf(qv[i], kv[j], attn[i][j]);
    }

    float p[4][4];
    #pragma unroll
    for (int i = 0; i < 4; ++i) {
        float m = fmaxf(fmaxf(attn[i][0], attn[i][1]), fmaxf(attn[i][2], attn[i][3]));
        float s = 0.f;
        #pragma unroll
        for (int j = 0; j < 4; ++j) { p[i][j] = __expf(attn[i][j] - m); s += p[i][j]; }
        float inv = 1.f / s;
        #pragma unroll
        for (int j = 0; j < 4; ++j) p[i][j] *= inv;
    }

    for (int d = 0; d < HDd; ++d) {
        int nsb = (hh * HDd + d) * 64 + w9;
        float vv[4];
        #pragma unroll
        for (int j = 0; j < 4; ++j)
            vv[j] = (float)V[kbase + (size_t)(nsb + j * 16) * Cc];
        #pragma unroll
        for (int i = 0; i < 4; ++i) {
            float o = fmaf(p[i][0], vv[0], fmaf(p[i][1], vv[1],
                      fmaf(p[i][2], vv[2], p[i][3] * vv[3])));
            out[((size_t)b * Ll + (size_t)i * WRES + w) * Cc + hh * HDd + d] = o;
        }
    }
}

// ============================================================
extern "C" void kernel_launch(void* const* d_in, const int* in_sizes, int n_in,
                              void* d_out, int out_size, void* d_ws, size_t ws_size,
                              hipStream_t stream)
{
    const float* s     = (const float*)d_in[0];
    const float* x     = (const float*)d_in[1];
    const float* s_pos = (const float*)d_in[2];
    const float* x_pos = (const float*)d_in[3];
    const float* Wq = (const float*)d_in[4];
    const float* bq = (const float*)d_in[5];
    const float* Wk = (const float*)d_in[6];
    const float* bk = (const float*)d_in[7];
    const float* Wv = (const float*)d_in[8];
    const float* bv = (const float*)d_in[9];
    const float* pw1  = (const float*)d_in[10];
    const float* pb1  = (const float*)d_in[11];
    const float* bn_g = (const float*)d_in[12];
    const float* bn_b = (const float*)d_in[13];
    const float* bn_m = (const float*)d_in[14];
    const float* bn_v = (const float*)d_in[15];
    const float* pw2  = (const float*)d_in[16];
    const float* pb2  = (const float*)d_in[17];
    float* out = (float*)d_out;

    const size_t M = (size_t)Bb * Ll;          // 65536 rows for the GEMMs
    char* ws = (char*)d_ws;
    size_t off = 0;
    __bf16* Qbf   = (__bf16*)(ws + off); off += M * Cc * sizeof(__bf16);
    __bf16* Kbf   = (__bf16*)(ws + off); off += M * Cc * sizeof(__bf16);
    __bf16* Vbf   = (__bf16*)(ws + off); off += M * Cc * sizeof(__bf16);
    float*  biasw = (float*) (ws + off); off += (size_t)NWw * NHh * 16 * sizeof(float);
    __bf16* WTq   = (__bf16*)(ws + off); off += (size_t)Cc * Cc * sizeof(__bf16);
    __bf16* WTk   = (__bf16*)(ws + off); off += (size_t)Cc * Cc * sizeof(__bf16);
    __bf16* WTv   = (__bf16*)(ws + off); off += (size_t)Cc * Cc * sizeof(__bf16);

    // weight prep (bf16, transposed to fragment-ready [n][k])
    dim3 wt((Cc * Cc) / 256);
    transpose_w_kernel<<<wt, 256, 0, stream>>>(Wq, WTq);
    transpose_w_kernel<<<wt, 256, 0, stream>>>(Wk, WTk);
    transpose_w_kernel<<<wt, 256, 0, stream>>>(Wv, WTv);

    bias_kernel<<<(NWw + 127) / 128, 128, 0, stream>>>(
        s_pos, x_pos, pw1, pb1, bn_g, bn_b, bn_m, bn_v, pw2, pb2, biasw);

    dim3 gg((unsigned)((M / BM) * (Cc / BN)));   // 4096 blocks
    gemm_proj_kernel<false><<<gg, 256, 0, stream>>>(x, WTq, bq, Qbf,
                                                    nullptr, nullptr, nullptr);
    gemm_proj_kernel<true ><<<gg, 256, 0, stream>>>(s, WTk, bk, Kbf,
                                                    WTv, bv, Vbf);

    attn_kernel<<<(NWw * NHh) / 256, 256, 0, stream>>>(Qbf, Kbf, Vbf, biasw, out);
}
